// CrossModalAttention_74002286510558
// MI455X (gfx1250) — compile-verified, matrix operand order checked by reading, not run on hardware
//
#include <hip/hip_runtime.h>

typedef unsigned short u16;
typedef __bf16 bf16_t;
typedef bf16_t v16bf __attribute__((ext_vector_type(16)));
typedef float  v8f   __attribute__((ext_vector_type(8)));
typedef u16    v4u16 __attribute__((ext_vector_type(4)));
typedef int    v4i   __attribute__((vector_size(16)));   // matches builtin param type

// ---- gfx1250 async global->LDS path (guarded; falls back to sync staging) ----
#if defined(__gfx1250__) && __has_builtin(__builtin_amdgcn_global_load_async_to_lds_b128) && __has_builtin(__builtin_amdgcn_s_wait_asynccnt)
#define HAVE_ASYNC_LDS 1
#else
#define HAVE_ASYNC_LDS 0
#endif

union Frag {
  v16bf v;
  uint4 q[2];
  u16   h[16];
};

__device__ __forceinline__ u16 bf16_rn(float f) {
  unsigned u = __float_as_uint(f);
  u += 0x7FFFu + ((u >> 16) & 1u);
  return (u16)(u >> 16);
}

__device__ __forceinline__ void load_frag_lds(Frag& fr, const u16* p) {
  fr.q[0] = *(const uint4*)(p);
  fr.q[1] = *(const uint4*)(p + 16);
}

__device__ __forceinline__ v8f wmma_bf16(const Frag& a, const Frag& b, v8f c) {
  return __builtin_amdgcn_wmma_f32_16x16x32_bf16(
      /*neg_a=*/false, a.v, /*neg_b=*/false, b.v,
      /*c_mod=*/(short)0, c, /*reuse_a=*/false, /*reuse_b=*/false);
}

// async copy of 16 bytes global -> LDS (per lane); no-op on fallback path
__device__ __forceinline__ void async_b128(const void* g, void* l) {
#if HAVE_ASYNC_LDS
  __builtin_amdgcn_global_load_async_to_lds_b128(
      (__attribute__((address_space(1))) v4i*)(unsigned long long)(__SIZE_TYPE__)g,
      (__attribute__((address_space(3))) v4i*)(unsigned)(__SIZE_TYPE__)l,
      0, 0);
#else
  (void)g; (void)l;
#endif
}

// ---------------------------------------------------------------------------
// GEMM: out[M,N] = A[M,K] @ B[K,N] + bias[N]
// A is fp32 or bf16 (A_BF16), out is bf16 or fp32 (OUT_BF16). B (weights) fp32.
// Block tile 128x64, 8 waves, each wave a 32x32 patch of 2x2 WMMA tiles.
// ---------------------------------------------------------------------------
#define KPAD 40

template <bool A_BF16, bool OUT_BF16>
__global__ __launch_bounds__(256)
void gemm_bias_kernel(const void* __restrict__ Aptr,
                      const float* __restrict__ Bw,
                      const float* __restrict__ bias,
                      void* __restrict__ outp,
                      int M, int N, int K)
{
  __shared__ u16 As[128][KPAD];
  __shared__ u16 Bs[64][KPAD];
  const int tid  = threadIdx.x;
  const int lane = tid & 31;
  const int w    = tid >> 5;
  const int hf   = lane >> 4;   // half-wave: selects K sub-pattern
  const int ln   = lane & 15;   // M row (A frag) / N col (B,D frags)
  const int row0 = blockIdx.y * 128;
  const int col0 = blockIdx.x * 64;
  const int wm   = (w >> 1) * 32;
  const int wn   = (w & 1) * 32;

  v8f acc[2][2] = {};

  #pragma unroll 1
  for (int kc = 0; kc < K; kc += 32) {
    // ---- stage A chunk (128 x 32) into As[m][k] (bf16) ----
    if (A_BF16) {
      const u16* Abf = (const u16*)Aptr;
      #pragma unroll
      for (int i = 0; i < 2; ++i) {
        int s = tid + i * 256;                 // 0..511
        int m = s >> 2;
        int kq = (s & 3) * 8;
        const u16* src = Abf + (size_t)(row0 + m) * K + kc + kq;
#if HAVE_ASYNC_LDS
        async_b128(src, &As[m][kq]);
#else
        uint4 d = *(const uint4*)src;
        *(uint4*)&As[m][kq] = d;
#endif
      }
    } else {
      const float* Af = (const float*)Aptr;
      #pragma unroll
      for (int i = 0; i < 4; ++i) {
        int s = tid + i * 256;                 // 0..1023
        int m = s >> 3;
        int kq = (s & 7) * 4;
        float4 f = *(const float4*)(Af + (size_t)(row0 + m) * K + kc + kq);
        v4u16 hh;
        hh[0] = bf16_rn(f.x); hh[1] = bf16_rn(f.y);
        hh[2] = bf16_rn(f.z); hh[3] = bf16_rn(f.w);
        *(v4u16*)&As[m][kq] = hh;
      }
    }
    // ---- stage B chunk (32 x 64) transposed into Bs[n][k] (bf16) ----
    #pragma unroll
    for (int i = 0; i < 2; ++i) {
      int s = tid + i * 256;                   // 0..511
      int k = s >> 4;
      int nq = (s & 15) * 4;
      float4 f = *(const float4*)(Bw + (size_t)(kc + k) * N + col0 + nq);
      Bs[nq + 0][k] = bf16_rn(f.x);
      Bs[nq + 1][k] = bf16_rn(f.y);
      Bs[nq + 2][k] = bf16_rn(f.z);
      Bs[nq + 3][k] = bf16_rn(f.w);
    }
#if HAVE_ASYNC_LDS
    if (A_BF16) __builtin_amdgcn_s_wait_asynccnt(0);
#endif
    __syncthreads();

    Frag a[2], bf[2];
    #pragma unroll
    for (int mt = 0; mt < 2; ++mt)
      load_frag_lds(a[mt], &As[wm + mt * 16 + ln][hf * 8]);
    #pragma unroll
    for (int nt = 0; nt < 2; ++nt)
      load_frag_lds(bf[nt], &Bs[wn + nt * 16 + ln][hf * 8]);
    #pragma unroll
    for (int mt = 0; mt < 2; ++mt)
      #pragma unroll
      for (int nt = 0; nt < 2; ++nt)
        acc[mt][nt] = wmma_bf16(a[mt], bf[nt], acc[mt][nt]);
    __syncthreads();
  }

  // epilogue: D layout -> lane: n=ln, vgpr v: m = v + hf*8
  #pragma unroll
  for (int mt = 0; mt < 2; ++mt) {
    #pragma unroll
    for (int nt = 0; nt < 2; ++nt) {
      int gn = col0 + wn + nt * 16 + ln;
      float bb = bias[gn];
      #pragma unroll
      for (int v = 0; v < 8; ++v) {
        int gm = row0 + wm + mt * 16 + v + hf * 8;
        float val = acc[mt][nt][v] + bb;
        if (OUT_BF16)
          ((u16*)outp)[(size_t)gm * N + gn] = bf16_rn(val);
        else
          ((float*)outp)[(size_t)gm * N + gn] = val;
      }
    }
  }
}

// ---------------------------------------------------------------------------
// Attention: per block = (b, h, 32-query tile). Sq=2048 Sk=576 Hd=256.
// K chunks are double-buffered and (when available) staged with async
// global->LDS copies overlapped with the WMMA score computation.
// ---------------------------------------------------------------------------
#define QPAD 264
#define VPAD 40
#define SPAD 584

struct AttnSmem {
  u16 Qs[32][QPAD];                 // Q tile  [q][d]
  union {
    u16 Ks[2][64][QPAD];            // K chunk [buf][sk][d]  (phase 1)
    u16 Vt[256][VPAD];              // V chunk transposed [d][sk] (phase 3)
  };
  float S[32][SPAD];                // scores / probabilities [q][sk]
};

__global__ __launch_bounds__(256)
void attn_kernel(const u16* __restrict__ Qbf, const u16* __restrict__ Kbf,
                 const u16* __restrict__ Vbf, u16* __restrict__ Ctx,
                 float* __restrict__ attn_mean,
                 const float* __restrict__ temperature)
{
  __shared__ AttnSmem sm;
  const int tid  = threadIdx.x;
  const int lane = tid & 31;
  const int w    = tid >> 5;
  const int hf   = lane >> 4;
  const int ln   = lane & 15;
  const int q0   = blockIdx.x * 32;
  const int h    = blockIdx.y;
  const int b    = blockIdx.z;
  const int Sq = 2048, Sk = 576, D = 4096, Hd = 256;

  const float scale = 0.0625f / temperature[0];   // 1/sqrt(256) / T

  // ---- stage Q tile [32][256] ----
  const u16* Qsrc = Qbf + ((size_t)(b * Sq + q0) * D) + h * Hd;
  #pragma unroll
  for (int i = 0; i < 4; ++i) {
    int s = tid + i * 256;                   // 32 rows x 32 groups of 8
    int m = s >> 5;
    int dq = (s & 31) * 8;
    const u16* src = Qsrc + (size_t)m * D + dq;
#if HAVE_ASYNC_LDS
    async_b128(src, &sm.Qs[m][dq]);
#else
    uint4 d = *(const uint4*)src;
    *(uint4*)&sm.Qs[m][dq] = d;
#endif
  }

  // per-thread stager for one 64-key chunk (8 x b128 per thread)
  auto stageK = [&](int skc, int buf) {
    const u16* Ksrc = Kbf + ((size_t)(b * Sk + skc * 64) * D) + h * Hd;
    #pragma unroll
    for (int i = 0; i < 8; ++i) {
      int s = tid + i * 256;                 // 64 rows x 32 groups
      int r = s >> 5;
      int dq = (s & 31) * 8;
      const u16* src = Ksrc + (size_t)r * D + dq;
#if HAVE_ASYNC_LDS
      async_b128(src, &sm.Ks[buf][r][dq]);
#else
      uint4 d = *(const uint4*)src;
      *(uint4*)&sm.Ks[buf][r][dq] = d;
#endif
    }
  };

  // ---- phase 1: scores = Q K^T * scale (double-buffered K chunks) ----
  const int mt = w & 1;
  const int nt = w >> 1;                      // 0..3
  stageK(0, 0);
  #pragma unroll 1
  for (int skc = 0; skc < Sk / 64; ++skc) {
    if (skc + 1 < Sk / 64) {
      stageK(skc + 1, (skc + 1) & 1);         // prefetch next chunk
#if HAVE_ASYNC_LDS
      __builtin_amdgcn_s_wait_asynccnt(8);    // current chunk (+Q) complete
#endif
    } else {
#if HAVE_ASYNC_LDS
      __builtin_amdgcn_s_wait_asynccnt(0);
#endif
    }
    __syncthreads();
    const u16 (*Kc)[QPAD] = sm.Ks[skc & 1];
    v8f sc = {};
    #pragma unroll
    for (int dc = 0; dc < 8; ++dc) {
      Frag a, bf;
      load_frag_lds(a,  &sm.Qs[mt * 16 + ln][dc * 32 + hf * 8]);
      load_frag_lds(bf, &Kc[nt * 16 + ln][dc * 32 + hf * 8]);
      sc = wmma_bf16(a, bf, sc);
    }
    #pragma unroll
    for (int v = 0; v < 8; ++v)
      sm.S[mt * 16 + v + hf * 8][skc * 64 + nt * 16 + ln] = sc[v] * scale;
    __syncthreads();
  }

  // ---- phase 2: softmax over Sk per row, accumulate attn_mean ----
  {
    int r = tid >> 3;
    int sub = tid & 7;
    float* Sr = sm.S[r];
    float mx = -3.0e38f;
    for (int c = sub; c < Sk; c += 8) mx = fmaxf(mx, Sr[c]);
    mx = fmaxf(mx, __shfl_xor(mx, 4, 8));
    mx = fmaxf(mx, __shfl_xor(mx, 2, 8));
    mx = fmaxf(mx, __shfl_xor(mx, 1, 8));
    float sum = 0.f;
    for (int c = sub; c < Sk; c += 8) {
      float e = __expf(Sr[c] - mx);
      Sr[c] = e;
      sum += e;
    }
    sum += __shfl_xor(sum, 4, 8);
    sum += __shfl_xor(sum, 2, 8);
    sum += __shfl_xor(sum, 1, 8);
    float inv = 1.f / sum;
    float* am = attn_mean + ((size_t)(b * Sq) + q0 + r) * Sk;
    for (int c = sub; c < Sk; c += 8) {
      float p = Sr[c] * inv;
      Sr[c] = p;
      atomicAdd(&am[c], p * 0.0625f);          // / NUM_HEADS
    }
  }

  // ---- phase 3: out = attn @ V ----
  const int dt0 = (w >> 1) * 4;
  v8f oacc[4] = {};
  const float* Sr = sm.S[mt * 16 + ln];
  #pragma unroll 1
  for (int skc = 0; skc < Sk / 32; ++skc) {
    __syncthreads();
    const u16* Vsrc = Vbf + ((size_t)(b * Sk + skc * 32) * D) + h * Hd;
    #pragma unroll
    for (int i = 0; i < 4; ++i) {
      int s = tid + i * 256;                 // 32 rows x 32 groups
      int sk = s >> 5;
      int dq = (s & 31) * 8;
      uint4 d = *(const uint4*)(Vsrc + (size_t)sk * D + dq);
      const u16* hh = (const u16*)&d;
      #pragma unroll
      for (int j = 0; j < 8; ++j)
        sm.Vt[dq + j][sk] = hh[j];
    }
    __syncthreads();
    Frag af;
    #pragma unroll
    for (int e = 0; e < 16; ++e) {
      int kk = e + ((e >= 8) ? 8 : 0) + hf * 8;
      af.h[e] = bf16_rn(Sr[skc * 32 + kk]);
    }
    #pragma unroll
    for (int j = 0; j < 4; ++j) {
      Frag vf;
      load_frag_lds(vf, &sm.Vt[(dt0 + j) * 16 + ln][hf * 8]);
      oacc[j] = wmma_bf16(af, vf, oacc[j]);
    }
  }

  // write bf16 context [B,Sq,D]
  #pragma unroll
  for (int j = 0; j < 4; ++j) {
    int gd = h * Hd + (dt0 + j) * 16 + ln;
    #pragma unroll
    for (int v = 0; v < 8; ++v) {
      int gq = q0 + mt * 16 + v + hf * 8;
      Ctx[((size_t)(b * Sq) + gq) * D + gd] = bf16_rn(oacc[j][v]);
    }
  }
}

__global__ void zero_kernel(float* __restrict__ p, int n) {
  int i = blockIdx.x * blockDim.x + threadIdx.x;
  if (i < n) p[i] = 0.f;
}

// ---------------------------------------------------------------------------
extern "C" void kernel_launch(void* const* d_in, const int* in_sizes, int n_in,
                              void* d_out, int out_size, void* d_ws, size_t ws_size,
                              hipStream_t stream) {
  (void)in_sizes; (void)n_in; (void)out_size; (void)ws_size;
  const float* query = (const float*)d_in[0];
  const float* key   = (const float*)d_in[1];
  const float* value = (const float*)d_in[2];
  const float* Wq = (const float*)d_in[3];
  const float* bq = (const float*)d_in[4];
  const float* Wk = (const float*)d_in[5];
  const float* bk = (const float*)d_in[6];
  const float* Wv = (const float*)d_in[7];
  const float* bv = (const float*)d_in[8];
  const float* Wo = (const float*)d_in[9];
  const float* bo = (const float*)d_in[10];
  const float* temp = (const float*)d_in[11];

  const int B = 4, Sq = 2048, Sk = 576, D = 4096;

  u16* Qbf = (u16*)d_ws;
  u16* Kbf = Qbf + (size_t)B * Sq * D;
  u16* Vbf = Kbf + (size_t)B * Sk * D;
  u16* Ctx = Vbf + (size_t)B * Sk * D;

  float* out = (float*)d_out;
  float* attn_mean = out + (size_t)B * Sq * D;

  // zero attn_mean (accumulated with atomics across heads)
  int am_n = B * Sq * Sk;
  zero_kernel<<<(am_n + 255) / 256, 256, 0, stream>>>(attn_mean, am_n);

  // projections (fp32 A, bf16 out)
  gemm_bias_kernel<false, true><<<dim3(D / 64, (B * Sq) / 128), 256, 0, stream>>>(
      query, Wq, bq, Qbf, B * Sq, D, D);
  gemm_bias_kernel<false, true><<<dim3(D / 64, (B * Sk) / 128), 256, 0, stream>>>(
      key, Wk, bk, Kbf, B * Sk, D, D);
  gemm_bias_kernel<false, true><<<dim3(D / 64, (B * Sk) / 128), 256, 0, stream>>>(
      value, Wv, bv, Vbf, B * Sk, D, D);

  // attention
  attn_kernel<<<dim3(Sq / 32, 16, B), 256, 0, stream>>>(
      Qbf, Kbf, Vbf, Ctx, attn_mean, temp);

  // output projection (bf16 A, fp32 out)
  gemm_bias_kernel<true, false><<<dim3(D / 64, (B * Sq) / 128), 256, 0, stream>>>(
      Ctx, Wo, bo, out, B * Sq, D, D);
}
